// Memory_trans_update_38079180046959
// MI455X (gfx1250) — matrix operand order repaired and checked by reading, not compiled, not run on hardware
//
#include <hip/hip_runtime.h>
#include <cstddef>

// ---------------- types / helpers ----------------
typedef __attribute__((ext_vector_type(16))) __bf16 bf16x16;
typedef __attribute__((ext_vector_type(8)))  float  f32x8;

union FragBF { uint4 u[2]; bf16x16 v; };

__device__ __forceinline__ unsigned short f32_to_bf16(float x) {
  unsigned u = __float_as_uint(x);
  u += 0x7FFFu + ((u >> 16) & 1u);          // round-to-nearest-even
  return (unsigned short)(u >> 16);
}
// order-preserving f32 <-> u32 (for unsigned atomicMax on floats)
__device__ __forceinline__ unsigned ford(float x) {
  unsigned u = __float_as_uint(x);
  return (u & 0x80000000u) ? ~u : (u | 0x80000000u);
}
__device__ __forceinline__ float funord(unsigned u) {
  unsigned b = (u & 0x80000000u) ? (u & 0x7FFFFFFFu) : ~u;
  return __uint_as_float(b);
}
__device__ __forceinline__ f32x8 wmma_bf16(bf16x16 a, bf16x16 b, f32x8 c) {
  // v_wmma_f32_16x16x32_bf16: (neg_a, A, neg_b, B, c_mod, C, reuse_a, reuse_b)
  return __builtin_amdgcn_wmma_f32_16x16x32_bf16(false, a, false, b, (short)0, c,
                                                 false, false);
}
__device__ __forceinline__ unsigned long long shfl_xor_u64_w16(unsigned long long x, int m) {
  unsigned lo = (unsigned)x, hi = (unsigned)(x >> 32);
  lo = __shfl_xor(lo, m, 16);
  hi = __shfl_xor(hi, m, 16);
  return ((unsigned long long)hi << 32) | (unsigned long long)lo;
}

// Problem sizes (fixed by reference)
#define NQ 32768      // 32*32*32 query rows
#define MKEY 512
#define DCH 256
#define SPAT 1024     // 32*32 spatial per batch
#define BSTRIDE 262144 // 256*1024 floats per batch

#define MB 32         // rows per workgroup in GEMM pass
#define LDA 264       // LDS row stride in bf16 elems (528B: 16B-aligned, conflict-free)

// ---------------- pass 0: init scratch + split keys to bf16 hi/lo ----------------
__global__ __launch_bounds__(256) void init_kernel(const float* __restrict__ keys,
                                                   float* __restrict__ upd,
                                                   unsigned* __restrict__ colmax,
                                                   unsigned short* __restrict__ khi,
                                                   unsigned short* __restrict__ klo) {
  int i = blockIdx.x * 256 + threadIdx.x;
  if (i < MKEY * DCH) {
    upd[i] = 0.0f;
    float k = keys[i];
    unsigned short h = f32_to_bf16(k);
    khi[i] = h;
    klo[i] = f32_to_bf16(k - __uint_as_float((unsigned)h << 16));
  }
  if (i < MKEY) colmax[i] = 0u;  // identity for ordered-max
}

// ---------------- pass 1: normalize q, WMMA GEMM, row argmax + col max ----------------
__global__ __launch_bounds__(256) void score_rowstats_kernel(
    const float* __restrict__ query,
    const unsigned short* __restrict__ khi, const unsigned short* __restrict__ klo,
    float* __restrict__ rowval, unsigned* __restrict__ rowidx,
    unsigned* __restrict__ colmax) {
  __shared__ __align__(16) unsigned short a_hi[MB * LDA];
  __shared__ __align__(16) unsigned short a_lo[MB * LDA];
  __shared__ unsigned long long rowkey[MB];
  __shared__ float invn[MB];
  __shared__ float partial[256];

  const int t = threadIdx.x;
  const int mbase = blockIdx.x * MB;

  // --- per-row inverse L2 norm over 256 channels ---
  {
    int r = t >> 3, part = t & 7;              // 32 rows x 8 partials
    int n = mbase + r;
    int b = n >> 10, s = n & 1023;
    const float* qp = query + (size_t)b * BSTRIDE + s;
    float ss = 0.0f;
    int c0 = part * 32;
    for (int c = c0; c < c0 + 32; ++c) { float x = qp[(size_t)c * SPAT]; ss += x * x; }
    partial[t] = ss;
  }
  __syncthreads();
  if (t < MB) {
    float ss = 0.0f;
    for (int p = 0; p < 8; ++p) ss += partial[t * 8 + p];
    invn[t] = 1.0f / fmaxf(sqrtf(ss), 1e-12f);
    rowkey[t] = 0ull;
  }
  __syncthreads();

  // --- stage normalized q as split bf16 (hi + lo) in LDS, row-major [row][K] ---
  {
    int r = t & 31, cg = t >> 5;               // lanes sweep rows -> coalesced
    int n = mbase + r;
    int b = n >> 10, s = n & 1023;
    const float* qp = query + (size_t)b * BSTRIDE + s;
    float inv = invn[r];
    for (int j = 0; j < 32; ++j) {
      int c = cg + 8 * j;
      float x = qp[(size_t)c * SPAT] * inv;
      unsigned short h = f32_to_bf16(x);
      a_hi[r * LDA + c] = h;
      a_lo[r * LDA + c] = f32_to_bf16(x - __uint_as_float((unsigned)h << 16));
    }
  }
  __syncthreads();

  const int lane = t & 31, wid = t >> 5;
  const int laneLo = lane & 15, half = lane >> 4;

  for (int nti = 0; nti < 4; ++nti) {          // uniform across waves
    const int ntg = wid * 4 + nti;             // 0..31 -> all 512 key rows
    const int col = ntg * 16 + laneLo;         // this lane's key row / score column

    f32x8 acc[2];
    for (int mt = 0; mt < 2; ++mt)
      for (int g = 0; g < 8; ++g) acc[mt][g] = 0.0f;

    for (int ks = 0; ks < 8; ++ks) {           // K = 256 in steps of 32
      const int kb = ks * 32;
      // B fragments: 32 contiguous bytes of bf16 K-values for this key row
      FragBF bh, bl;
      const uint4* bhp = (const uint4*)(khi + (size_t)col * DCH + kb + 16 * half);
      const uint4* blp = (const uint4*)(klo + (size_t)col * DCH + kb + 16 * half);
      bh.u[0] = bhp[0]; bh.u[1] = bhp[1];
      bl.u[0] = blp[0]; bl.u[1] = blp[1];
      for (int mt = 0; mt < 2; ++mt) {
        const int row = mt * 16 + laneLo;
        // A fragment: 16 contiguous K per half (matches B) -> two 16B chunks
        FragBF ah, al;
        ah.u[0] = *(const uint4*)(a_hi + row * LDA + kb + 16 * half);
        ah.u[1] = *(const uint4*)(a_hi + row * LDA + kb + 16 * half + 8);
        al.u[0] = *(const uint4*)(a_lo + row * LDA + kb + 16 * half);
        al.u[1] = *(const uint4*)(a_lo + row * LDA + kb + 16 * half + 8);
        // split-bf16 f32 emulation: qh*kh + qh*kl + ql*kh
        acc[mt] = wmma_bf16(al.v, bh.v, acc[mt]);
        acc[mt] = wmma_bf16(ah.v, bl.v, acc[mt]);
        acc[mt] = wmma_bf16(ah.v, bh.v, acc[mt]);
      }
    }

    // --- reductions on this wave's 32x64 score tile ---
    float cmax = -3.4e38f;
    for (int mt = 0; mt < 2; ++mt) {
      for (int g = 0; g < 8; ++g) {
        float v = acc[mt][g];
        cmax = fmaxf(cmax, v);
        // packed key: bigger score wins, tie -> smaller column (jnp.argmax order)
        unsigned long long key =
            ((unsigned long long)ford(v) << 32) |
            (unsigned long long)(0xFFFFFFFFu - (unsigned)col);
        for (int o = 1; o < 16; o <<= 1) {
          unsigned long long other = shfl_xor_u64_w16(key, o);
          if (other > key) key = other;
        }
        if (laneLo == 0) {
          int row = mt * 16 + 8 * half + g;    // C layout: M = g + 8*(lane>>4)
          atomicMax(&rowkey[row], key);        // ds_max_u64
        }
      }
    }
    // column max: combine lane halves (same column, disjoint rows)
    float oth = __shfl_xor(cmax, 16, 32);
    cmax = fmaxf(cmax, oth);
    if (half == 0) atomicMax(colmax + col, ford(cmax));
  }
  __syncthreads();

  if (t < MB) {
    unsigned long long key = rowkey[t];
    rowval[mbase + t] = funord((unsigned)(key >> 32));
    rowidx[mbase + t] = 0xFFFFFFFFu - (unsigned)(key & 0xFFFFFFFFull);
  }
}

// ---------------- pass 2: w = exp(s - colmax); scatter-add w*value ----------------
__global__ __launch_bounds__(256) void scatter_kernel(const float* __restrict__ value,
                                                      const float* __restrict__ rowval,
                                                      const unsigned* __restrict__ rowidx,
                                                      const unsigned* __restrict__ colmax,
                                                      float* __restrict__ upd) {
  int t = threadIdx.x;
  int n = blockIdx.x * 256 + t;                // 256 consecutive rows share batch b
  int b = n >> 10, s = n & 1023;
  unsigned idx = rowidx[n];
  float w = expf(rowval[n] - funord(colmax[idx]));
  const float* vp = value + (size_t)b * BSTRIDE + s;
  float* up = upd + (size_t)idx * DCH;
  for (int c = 0; c < DCH; ++c) {              // lanes sweep s -> coalesced loads
    atomicAdd(up + c, w * vp[(size_t)c * SPAT]);
  }
}

// ---------------- pass 3: out = l2norm(upd + keys) per row ----------------
__global__ __launch_bounds__(256) void finalize_kernel(const float* __restrict__ keys,
                                                       const float* __restrict__ upd,
                                                       float* __restrict__ out) {
  __shared__ float wsum[8];
  int r = blockIdx.x, c = threadIdx.x;
  float x = upd[r * DCH + c] + keys[r * DCH + c];
  float ss = x * x;
  for (int o = 16; o >= 1; o >>= 1) ss += __shfl_xor(ss, o, 32);
  int lane = c & 31, wid = c >> 5;
  if (lane == 0) wsum[wid] = ss;
  __syncthreads();
  if (c == 0) {
    float s = 0.0f;
    for (int i = 0; i < 8; ++i) s += wsum[i];
    wsum[0] = 1.0f / fmaxf(sqrtf(s), 1e-12f);
  }
  __syncthreads();
  out[r * DCH + c] = x * wsum[0];
}

// ---------------- launch ----------------
extern "C" void kernel_launch(void* const* d_in, const int* in_sizes, int n_in,
                              void* d_out, int out_size, void* d_ws, size_t ws_size,
                              hipStream_t stream) {
  const float* keys  = (const float*)d_in[0];  // [512,256]
  const float* query = (const float*)d_in[1];  // [32,256,32,32]
  const float* value = (const float*)d_in[2];  // [32,256,32,32]
  float* out = (float*)d_out;                  // [512,256]

  char* ws = (char*)d_ws;
  float*          upd    = (float*)(ws);                    // 512*256 f32 = 524288 B
  unsigned*       colmax = (unsigned*)(ws + 524288);        // 512 u32   =   2048 B
  float*          rowval = (float*)(ws + 526336);           // 32768 f32 = 131072 B
  unsigned*       rowidx = (unsigned*)(ws + 657408);        // 32768 u32 = 131072 B
  unsigned short* khi    = (unsigned short*)(ws + 788480);  // 512*256 bf16 = 262144 B
  unsigned short* klo    = (unsigned short*)(ws + 1050624); // 512*256 bf16 = 262144 B

  init_kernel<<<512, 256, 0, stream>>>(keys, upd, colmax, khi, klo);
  score_rowstats_kernel<<<NQ / MB, 256, 0, stream>>>(query, khi, klo,
                                                     rowval, rowidx, colmax);
  scatter_kernel<<<NQ / 256, 256, 0, stream>>>(value, rowval, rowidx, colmax, upd);
  finalize_kernel<<<MKEY, 256, 0, stream>>>(keys, upd, out);
}
